// CrossTransformer_11665131176458
// MI455X (gfx1250) — matrix operand order repaired and checked by reading, not compiled, not run on hardware
//
#include <hip/hip_runtime.h>
#include <hip/hip_bf16.h>

#define DIM   1024
#define DEPTH 3
#define HEADS 8
#define HD    128
#define MLP   4096
#define NTOK  1026            // 2 + 1024
#define BATCH 8
#define MPAD  8320            // 65*128 >= BATCH*NTOK = 8208
#define EPSV  1e-5f

typedef __bf16 bh16;
typedef __attribute__((ext_vector_type(16))) __bf16 v16bf;
typedef __attribute__((ext_vector_type(8)))  __bf16 bfx8;
typedef __attribute__((ext_vector_type(8)))  float  v8f;

union BFrag { v16bf v; bfx8 h[2]; };

__device__ __forceinline__ v8f wmma_bf16(v16bf a, v16bf b, v8f c) {
  // v_wmma_f32_16x16x32_bf16: D = A(16x32) * B(32x16) + C(16x16 f32)
  return __builtin_amdgcn_wmma_f32_16x16x32_bf16(false, a, false, b, (short)0, c, false, false);
}
__device__ __forceinline__ float gelu_exact(float x) {
  return 0.5f * x * (1.0f + erff(x * 0.70710678118654752f));
}
__device__ __forceinline__ int imin(int a, int b) { return a < b ? a : b; }
__device__ __forceinline__ float rowmax16(float t) {
  t = fmaxf(t, __shfl_xor(t, 1, 32));
  t = fmaxf(t, __shfl_xor(t, 2, 32));
  t = fmaxf(t, __shfl_xor(t, 4, 32));
  t = fmaxf(t, __shfl_xor(t, 8, 32));
  return t;
}
__device__ __forceinline__ float rowsum16(float t) {
  t += __shfl_xor(t, 1, 32);
  t += __shfl_xor(t, 2, 32);
  t += __shfl_xor(t, 4, 32);
  t += __shfl_xor(t, 8, 32);
  return t;
}

// gfx1250 async copy global -> LDS (ASYNCcnt-tracked, bypasses VGPRs)
__device__ __forceinline__ void async_load_b128(unsigned lds_off, const void* gptr) {
  unsigned long long ga = (unsigned long long)(uintptr_t)gptr;
  asm volatile("global_load_async_to_lds_b128 %0, %1, off"
               :: "v"(lds_off), "v"(ga) : "memory");
}
__device__ __forceinline__ void wait_async_le4() {
  asm volatile("s_wait_asynccnt 0x4" ::: "memory");
}
__device__ __forceinline__ void wait_async_0() {
  asm volatile("s_wait_asynccnt 0x0" ::: "memory");
}
// generic pointer low 32 bits == LDS byte offset (ISA flat aperture rules)
__device__ __forceinline__ unsigned lds_off_of(const void* p) {
  return (unsigned)(uintptr_t)p;
}

// ---------------- weight convert: fp32 [L,K,N] -> bf16 transposed [L,N,K] ----
__global__ void convw_kernel(const float* __restrict__ src, bh16* __restrict__ dst,
                             int K, int N) {
  int l = blockIdx.z;
  size_t i = (size_t)blockIdx.x * 256 + threadIdx.x;
  size_t tot = (size_t)K * N;
  if (i >= tot) return;
  int k = (int)(i / N), n = (int)(i % N);
  dst[((size_t)l * N + n) * K + k] = (bh16)src[((size_t)l * K + k) * N + n];
}

// ---------------- build residual stream c (fp32), zero-padded rows ----------
__global__ void build_c_kernel(const float* __restrict__ x, const float* __restrict__ z,
                               const float* __restrict__ zz, float* __restrict__ c) {
  size_t i = (size_t)blockIdx.x * 256 + threadIdx.x;
  if (i >= (size_t)MPAD * DIM) return;
  int row = (int)(i >> 10), d = (int)(i & 1023);
  float v = 0.0f;
  if (row < BATCH * NTOK) {
    int b = row / NTOK, t = row % NTOK;
    if (t == 0)      v = x[b * DIM + d];
    else if (t == 1) v = z[b * DIM + d];
    else             v = zz[((size_t)b * 1024 + (t - 2)) * DIM + d];
  }
  c[i] = v;
}

// ---------------- LayerNorm: c fp32 row -> h bf16 row -----------------------
__global__ __launch_bounds__(256) void ln_kernel(const float* __restrict__ c,
                                                 const float* __restrict__ w,
                                                 const float* __restrict__ bb,
                                                 bh16* __restrict__ h) {
  __shared__ float s1[256], s2[256];
  int row = blockIdx.x, tid = threadIdx.x;
  const float* cr = c + (size_t)row * DIM;
  float v[4], s = 0.f, ss = 0.f;
#pragma unroll
  for (int j = 0; j < 4; ++j) { v[j] = cr[tid + 256 * j]; s += v[j]; ss += v[j] * v[j]; }
  s1[tid] = s; s2[tid] = ss;
  __syncthreads();
  for (int st = 128; st > 0; st >>= 1) {
    if (tid < st) { s1[tid] += s1[tid + st]; s2[tid] += s2[tid + st]; }
    __syncthreads();
  }
  float mu = s1[0] * (1.0f / DIM);
  float var = s2[0] * (1.0f / DIM) - mu * mu;
  float r = rsqrtf(var + EPSV);
#pragma unroll
  for (int j = 0; j < 4; ++j) {
    int d = tid + 256 * j;
    h[(size_t)row * DIM + d] = (bh16)((v[j] - mu) * r * w[d] + bb[d]);
  }
}

// ---------------- WMMA GEMM with async-to-LDS double buffering --------------
// A[M,K] bf16 x Wt[N,K] bf16 -> 128x128 block tile, wave = 64x32.
// mode 0: outb = A*W                  (bf16)
// mode 1: cio  = A*W + bias + cio     (fp32 residual, in-place)
// mode 2: outb = gelu(A*W + bias)     (bf16)
__global__ __launch_bounds__(256) void gemm_wmma(const bh16* __restrict__ A,
                                                 const bh16* __restrict__ Wt,
                                                 const float* __restrict__ bias,
                                                 float* __restrict__ cio,
                                                 bh16* __restrict__ outb,
                                                 int M, int N, int K, int mode) {
  __shared__ __attribute__((aligned(16))) bh16 sA[2][128 * 32];
  __shared__ __attribute__((aligned(16))) bh16 sB[2][128 * 32];

  int tid = threadIdx.x;
  int lane = tid & 31, wave = tid >> 5;
  int waveM = wave & 1, waveN = wave >> 1;
  int lmod = lane & 15, lh = lane >> 4;
  int rowBlk = blockIdx.y * 128;
  int colBlk = blockIdx.x * 128;

  // staging: 512 16B chunks per tile, 2 chunks of A + 2 of B per thread/stage.
  // chunk c: row = c>>2, sub = c&3 (row stride 32 elems = 64B = 4 chunks);
  // LDS offset = c*16 bytes (row-major layout matches chunk order).
  int c0 = tid * 2, c1 = tid * 2 + 1;
  const bh16* gA0 = A + (size_t)(rowBlk + (c0 >> 2)) * K + (c0 & 3) * 8;
  const bh16* gA1 = A + (size_t)(rowBlk + (c1 >> 2)) * K + (c1 & 3) * 8;
  const bh16* gB0 = Wt + (size_t)(colBlk + (c0 >> 2)) * K + (c0 & 3) * 8;
  const bh16* gB1 = Wt + (size_t)(colBlk + (c1 >> 2)) * K + (c1 & 3) * 8;
  unsigned ldsA = lds_off_of(&sA[0][0]);
  unsigned ldsB = lds_off_of(&sB[0][0]);
  const unsigned bufB = 128 * 32 * 2;  // 8192 bytes per buffer

  v8f acc[4][2];
#pragma unroll
  for (int mt = 0; mt < 4; ++mt)
#pragma unroll
    for (int nt = 0; nt < 2; ++nt)
#pragma unroll
      for (int e = 0; e < 8; ++e) acc[mt][nt][e] = 0.0f;

  int nsteps = K >> 5;
  // prologue: stage 0 into buffer 0 (4 async instrs -> ASYNCcnt 4 per wave)
  async_load_b128(ldsA + c0 * 16, gA0);
  async_load_b128(ldsA + c1 * 16, gA1);
  async_load_b128(ldsB + c0 * 16, gB0);
  async_load_b128(ldsB + c1 * 16, gB1);

#pragma unroll 1
  for (int i = 0; i < nsteps; ++i) {
    int cur = i & 1;
    if (i + 1 < nsteps) {
      int k1 = (i + 1) * 32;
      unsigned bo = (1 - cur) * bufB;
      async_load_b128(ldsA + bo + c0 * 16, gA0 + k1);
      async_load_b128(ldsA + bo + c1 * 16, gA1 + k1);
      async_load_b128(ldsB + bo + c0 * 16, gB0 + k1);
      async_load_b128(ldsB + bo + c1 * 16, gB1 + k1);
      wait_async_le4();   // stage i complete (async loads retire in order)
    } else {
      wait_async_0();
    }
    __syncthreads();      // every wave waited its own asyncs -> tile visible

    const bh16* tA = &sA[cur][0];
    const bh16* tB = &sB[cur][0];
    BFrag af[4], bfg[2];
#pragma unroll
    for (int mt = 0; mt < 4; ++mt) {
      const bfx8* p = (const bfx8*)(tA + (waveM * 64 + mt * 16 + lmod) * 32 + lh * 8);
      af[mt].h[0] = p[0];   // K = lh*8 .. +7
      af[mt].h[1] = p[2];   // K = lh*8 + 16 .. +23
    }
#pragma unroll
    for (int nt = 0; nt < 2; ++nt) {
      const bfx8* p = (const bfx8*)(tB + (waveN * 32 + nt * 16 + lmod) * 32 + lh * 8);
      bfg[nt].h[0] = p[0];  // K = lh*8 .. +7
      bfg[nt].h[1] = p[2];  // K = lh*8 + 16 .. +23
    }
#pragma unroll
    for (int mt = 0; mt < 4; ++mt)
#pragma unroll
      for (int nt = 0; nt < 2; ++nt)
        acc[mt][nt] = wmma_bf16(af[mt].v, bfg[nt].v, acc[mt][nt]);
    __syncthreads();      // all reads of buf[cur] done before it is restaged
  }

#pragma unroll
  for (int mt = 0; mt < 4; ++mt)
#pragma unroll
    for (int nt = 0; nt < 2; ++nt) {
      int col = colBlk + waveN * 32 + nt * 16 + lmod;
      float bv = bias ? bias[col] : 0.0f;
#pragma unroll
      for (int e = 0; e < 8; ++e) {
        int row = rowBlk + waveM * 64 + mt * 16 + e + lh * 8;
        size_t idx = (size_t)row * N + col;
        float v = acc[mt][nt][e] + bv;
        if (mode == 1)      cio[idx] = v + cio[idx];
        else if (mode == 2) outb[idx] = (bh16)gelu_exact(v);
        else                outb[idx] = (bh16)v;
      }
    }
}

// ---------------- Flash-style attention with WMMA + async K staging ---------
// qkv: bf16 rows [b*NTOK + t, 3*DIM]; o: bf16 rows [b*NTOK + t, DIM]
__global__ __launch_bounds__(128) void attn_wmma(const bh16* __restrict__ qkv,
                                                 bh16* __restrict__ o) {
  __shared__ __attribute__((aligned(16))) bh16 sK[32 * 128];      // [key][d]
  __shared__ __attribute__((aligned(16))) bh16 sVT[HD * 32];      // V^T: [d][key]
  __shared__ __attribute__((aligned(16))) bh16 sP[4][16 * 32];    // per-wave P tile

  int lane = threadIdx.x & 31, wave = threadIdx.x >> 5;
  int lmod = lane & 15, lh = lane >> 4;
  int bh = blockIdx.y, b = bh >> 3, hh = bh & 7;
  int qbase = blockIdx.x * 64 + wave * 16;
  const bh16* base = qkv + (size_t)b * NTOK * (3 * DIM);
  unsigned ldsK = lds_off_of(&sK[0]);
  const float scale = 0.03125f;  // 1024^-0.5

  // resident Q fragments (16 rows x 128 head-dim = 4 A-frags)
  BFrag qf[4];
  {
    int t = imin(qbase + lmod, NTOK - 1);
    const bfx8* p = (const bfx8*)(base + (size_t)t * (3 * DIM) + hh * HD + lh * 8);
#pragma unroll
    for (int d = 0; d < 4; ++d) { qf[d].h[0] = p[d * 4]; qf[d].h[1] = p[d * 4 + 2]; }
  }

  float m_run[8], l_run[8];
  v8f oacc[8];
#pragma unroll
  for (int e = 0; e < 8; ++e) { m_run[e] = -1e30f; l_run[e] = 0.0f; }
#pragma unroll
  for (int j = 0; j < 8; ++j)
#pragma unroll
    for (int e = 0; e < 8; ++e) oacc[j][e] = 0.0f;

#pragma unroll 1
  for (int kt = 0; kt < NTOK; kt += 32) {
    __syncthreads();  // prior readers of sK/sVT/sP done before restaging
    // async-stage K tile [32 keys][128 d]: 512 chunks / 128 threads
#pragma unroll
    for (int it = 0; it < 4; ++it) {
      int chunk = threadIdx.x + it * 128;
      int key = chunk >> 4, dc = chunk & 15;
      int krow = imin(kt + key, NTOK - 1);
      async_load_b128(ldsK + chunk * 16,
                      base + (size_t)krow * (3 * DIM) + DIM + hh * HD + dc * 8);
    }
    // stage V^T tile: 32 keys x 128 d -> sVT[d][key] (register transpose)
#pragma unroll
    for (int it = 0; it < 4; ++it) {
      int chunk = threadIdx.x + it * 128;
      int key = chunk >> 4, dc = chunk & 15;
      int krow = imin(kt + key, NTOK - 1);
      bfx8 vv = *(const bfx8*)(base + (size_t)krow * (3 * DIM) + 2 * DIM + hh * HD + dc * 8);
#pragma unroll
      for (int dd = 0; dd < 8; ++dd) sVT[(dc * 8 + dd) * 32 + key] = vv[dd];
    }
    wait_async_0();
    __syncthreads();  // sK + sVT visible to all waves

    // S = Q * K^T for 32 keys (two 16-col tiles), K frags from LDS
    v8f s0, s1;
#pragma unroll
    for (int e = 0; e < 8; ++e) { s0[e] = 0.0f; s1[e] = 0.0f; }
#pragma unroll
    for (int d = 0; d < 4; ++d) {
      BFrag kb0, kb1;
      const bfx8* p0 = (const bfx8*)(&sK[lmod * 128 + d * 32 + lh * 8]);
      const bfx8* p1 = (const bfx8*)(&sK[(16 + lmod) * 128 + d * 32 + lh * 8]);
      kb0.h[0] = p0[0]; kb0.h[1] = p0[2];
      kb1.h[0] = p1[0]; kb1.h[1] = p1[2];
      s0 = wmma_bf16(qf[d].v, kb0.v, s0);
      s1 = wmma_bf16(qf[d].v, kb1.v, s1);
    }

    // online softmax
    float msk0 = (kt + lmod < NTOK) ? 0.0f : -1e30f;
    float msk1 = (kt + 16 + lmod < NTOK) ? 0.0f : -1e30f;
#pragma unroll
    for (int e = 0; e < 8; ++e) {
      float a0 = s0[e] * scale + msk0;
      float a1 = s1[e] * scale + msk1;
      float tmax = rowmax16(fmaxf(a0, a1));
      float mnew = fmaxf(m_run[e], tmax);
      float alpha = __expf(m_run[e] - mnew);
      m_run[e] = mnew;
      float p0e = __expf(a0 - mnew);
      float p1e = __expf(a1 - mnew);
      l_run[e] = l_run[e] * alpha + rowsum16(p0e + p1e);
#pragma unroll
      for (int j = 0; j < 8; ++j) oacc[j][e] *= alpha;
      int r = e + lh * 8;
      sP[wave][r * 32 + lmod] = (bh16)p0e;
      sP[wave][r * 32 + 16 + lmod] = (bh16)p1e;
    }
    __syncthreads();  // sP visible

    // P (A-frag) and V^T (B-frags) from LDS; O += P * V
    BFrag pa;
    {
      const bfx8* pp = (const bfx8*)&sP[wave][lmod * 32 + lh * 8];
      pa.h[0] = pp[0];
      pa.h[1] = pp[2];
    }
#pragma unroll
    for (int j = 0; j < 8; ++j) {
      BFrag vb;
      const bfx8* pv = (const bfx8*)&sVT[(j * 16 + lmod) * 32 + lh * 8];
      vb.h[0] = pv[0]; vb.h[1] = pv[2];
      oacc[j] = wmma_bf16(pa.v, vb.v, oacc[j]);
    }
  }

  // write O / l
#pragma unroll
  for (int j = 0; j < 8; ++j) {
    int col = hh * HD + j * 16 + lmod;
#pragma unroll
    for (int e = 0; e < 8; ++e) {
      int t = qbase + e + lh * 8;
      if (t < NTOK) {
        float ov = oacc[j][e] / l_run[e];
        o[((size_t)b * NTOK + t) * DIM + col] = (bh16)ov;
      }
    }
  }
}

// ---------------- extract rows 0,1 of each batch ----------------------------
__global__ void extract_kernel(const float* __restrict__ c, float* __restrict__ out) {
  int i = blockIdx.x * 256 + threadIdx.x;  // 16384 total
  int j = i >> 13;
  int rem = i & 8191;
  int b = rem >> 10, d = rem & 1023;
  out[i] = c[((size_t)(b * NTOK + j)) * DIM + d];
}

extern "C" void kernel_launch(void* const* d_in, const int* in_sizes, int n_in,
                              void* d_out, int out_size, void* d_ws, size_t ws_size,
                              hipStream_t stream) {
  (void)in_sizes; (void)n_in; (void)out_size; (void)ws_size;
  const float* x    = (const float*)d_in[0];
  const float* z    = (const float*)d_in[1];
  const float* zz   = (const float*)d_in[2];
  const float* ln1w = (const float*)d_in[3];
  const float* ln1b = (const float*)d_in[4];
  const float* qkvw = (const float*)d_in[5];
  const float* outw = (const float*)d_in[6];
  const float* outb = (const float*)d_in[7];
  const float* ln2w = (const float*)d_in[8];
  const float* ln2b = (const float*)d_in[9];
  const float* ff1w = (const float*)d_in[10];
  const float* ff1b = (const float*)d_in[11];
  const float* ff2w = (const float*)d_in[12];
  const float* ff2b = (const float*)d_in[13];

  char* wsp = (char*)d_ws;
  auto carve = [&](size_t bytes) { char* p = wsp; wsp += (bytes + 255) & ~(size_t)255; return p; };
  float* c    = (float*)carve((size_t)MPAD * DIM * 4);
  bh16* h     = (bh16*)carve((size_t)MPAD * DIM * 2);
  bh16* qkv   = (bh16*)carve((size_t)MPAD * 3 * DIM * 2);
  bh16* ob    = (bh16*)carve((size_t)MPAD * DIM * 2);
  bh16* u     = (bh16*)carve((size_t)MPAD * MLP * 2);
  bh16* WqkvT = (bh16*)carve((size_t)DEPTH * 3 * DIM * DIM * 2);
  bh16* WoutT = (bh16*)carve((size_t)DEPTH * DIM * DIM * 2);
  bh16* Wff1T = (bh16*)carve((size_t)DEPTH * MLP * DIM * 2);
  bh16* Wff2T = (bh16*)carve((size_t)DEPTH * DIM * MLP * 2);

  dim3 blk256(256);
  convw_kernel<<<dim3((DIM * 3 * DIM + 255) / 256, 1, DEPTH), blk256, 0, stream>>>(qkvw, WqkvT, DIM, 3 * DIM);
  convw_kernel<<<dim3((DIM * DIM + 255) / 256, 1, DEPTH), blk256, 0, stream>>>(outw, WoutT, DIM, DIM);
  convw_kernel<<<dim3((DIM * MLP + 255) / 256, 1, DEPTH), blk256, 0, stream>>>(ff1w, Wff1T, DIM, MLP);
  convw_kernel<<<dim3((MLP * DIM + 255) / 256, 1, DEPTH), blk256, 0, stream>>>(ff2w, Wff2T, MLP, DIM);
  build_c_kernel<<<dim3(MPAD * DIM / 256), blk256, 0, stream>>>(x, z, zz, c);

  for (int i = 0; i < DEPTH; ++i) {
    ln_kernel<<<dim3(MPAD), blk256, 0, stream>>>(c, ln1w + i * DIM, ln1b + i * DIM, h);
    gemm_wmma<<<dim3(3 * DIM / 128, MPAD / 128), blk256, 0, stream>>>(
        h, WqkvT + (size_t)i * 3 * DIM * DIM, nullptr, nullptr, qkv, MPAD, 3 * DIM, DIM, 0);
    attn_wmma<<<dim3((NTOK + 63) / 64, BATCH * HEADS), dim3(128), 0, stream>>>(qkv, ob);
    gemm_wmma<<<dim3(DIM / 128, MPAD / 128), blk256, 0, stream>>>(
        ob, WoutT + (size_t)i * DIM * DIM, outb + i * DIM, c, nullptr, MPAD, DIM, DIM, 1);
    ln_kernel<<<dim3(MPAD), blk256, 0, stream>>>(c, ln2w + i * DIM, ln2b + i * DIM, h);
    gemm_wmma<<<dim3(MLP / 128, MPAD / 128), blk256, 0, stream>>>(
        h, Wff1T + (size_t)i * MLP * DIM, ff1b + i * MLP, nullptr, u, MPAD, MLP, DIM, 2);
    gemm_wmma<<<dim3(DIM / 128, MPAD / 128), blk256, 0, stream>>>(
        u, Wff2T + (size_t)i * DIM * MLP, ff2b + i * DIM, c, nullptr, MPAD, DIM, MLP, 1);
  }
  extract_kernel<<<dim3(64), blk256, 0, stream>>>(c, (float*)d_out);
}
